// PlanetoidGAT_15762529976324
// MI455X (gfx1250) — compile-verified, matrix operand order checked by reading, not run on hardware
//
#include <hip/hip_runtime.h>
#include <hip/hip_bf16.h>
#include <math.h>

typedef __attribute__((ext_vector_type(16))) __bf16 v16bf;
typedef __attribute__((ext_vector_type(8)))  float  v8f;

#define NN   50000
#define NE   800000
#define DIN_ 256
#define H_   2
#define DOUT_ 64

// ---------------------------------------------------------------------------
// float atomic max via sign-split int trick (works for mixed signs, init -inf)
// ---------------------------------------------------------------------------
__device__ __forceinline__ void atomicMaxF(float* addr, float val) {
    if (val >= 0.0f) {
        atomicMax((int*)addr, __float_as_int(val));
    } else {
        atomicMin((unsigned int*)addr, __float_as_uint(val));
    }
}

// ---------------------------------------------------------------------------
// Kernel 0: pack W[h][k][o] (fp32) into CDNA5 WMMA bf16 B-operand layout,
// split into hi/lo bf16 for near-fp32 accuracy.
// Layout per (h, ktile, ntile, lane): 16 contiguous bf16 (32B).
// lane<16: K = kb+{0..7, 16..23}, kb=0 ; lane>=16: kb=8. n = ntile*16+(lane&15)
// ---------------------------------------------------------------------------
__global__ void gat_pack_w(const float* __restrict__ W,
                           __bf16* __restrict__ whi, __bf16* __restrict__ wlo) {
    int t = blockIdx.x * 256 + threadIdx.x;        // (((h*8+kt)*4+j)*32+lane)
    if (t >= H_ * 8 * 4 * 32) return;
    int lane = t & 31;
    int j    = (t >> 5) & 3;
    int kt   = (t >> 7) & 7;
    int h    = t >> 10;
    int n    = j * 16 + (lane & 15);
    int kb   = (lane < 16) ? 0 : 8;
    int k0   = kt * 32;
    __bf16* dh = whi + (size_t)t * 16;
    __bf16* dl = wlo + (size_t)t * 16;
#pragma unroll
    for (int i = 0; i < 16; ++i) {
        int kl = (i < 8) ? (kb + i) : (kb + 8 + i);   // {kb..kb+7, kb+16..kb+23}
        float w = W[((size_t)h * DIN_ + (k0 + kl)) * DOUT_ + n];
        __bf16 hi = (__bf16)w;
        dh[i] = hi;
        dl[i] = (__bf16)(w - (float)hi);
    }
}

// ---------------------------------------------------------------------------
// Kernel 1: feats[h][n][o] = X[n,:] @ W[h] + b[h]  via v_wmma_f32_16x16x32_bf16
// Block 256 = 8 waves; wave computes 16 rows x 64 cols; grid (ceil(N/128), H).
// Split-bf16: acc += Ahi*Bhi + Ahi*Blo + Alo*Bhi  (~fp32 accuracy).
// ---------------------------------------------------------------------------
__global__ __launch_bounds__(256)
void gat_gemm(const float* __restrict__ X, const __bf16* __restrict__ whi,
              const __bf16* __restrict__ wlo, const float* __restrict__ bias,
              float* __restrict__ feats) {
    const int h    = blockIdx.y;
    const int wave = threadIdx.x >> 5;
    const int lane = threadIdx.x & 31;
    const int rbase = blockIdx.x * 128 + wave * 16;
    const int mrow  = lane & 15;
    const int kb    = (lane < 16) ? 0 : 8;

    int r = rbase + mrow;
    if (r > NN - 1) r = NN - 1;                      // clamp (guard at store)
    const float* ap0 = X + (size_t)r * DIN_ + kb;

    const __bf16* bhbase = whi + (size_t)h * (8 * 4 * 32 * 16) + (size_t)lane * 16;
    const __bf16* blbase = wlo + (size_t)h * (8 * 4 * 32 * 16) + (size_t)lane * 16;

    v8f acc[4] = {};

    for (int t = 0; t < 8; ++t) {                    // K loop, 32 per step
        const float* ap = ap0 + t * 32;
        float4 f0 = *(const float4*)(ap);
        float4 f1 = *(const float4*)(ap + 4);
        float4 f2 = *(const float4*)(ap + 16);
        float4 f3 = *(const float4*)(ap + 20);
        float a[16] = {f0.x, f0.y, f0.z, f0.w, f1.x, f1.y, f1.z, f1.w,
                       f2.x, f2.y, f2.z, f2.w, f3.x, f3.y, f3.z, f3.w};
        v16bf ahi, alo;
#pragma unroll
        for (int i = 0; i < 16; ++i) {
            __bf16 hi = (__bf16)a[i];
            ahi[i] = hi;
            alo[i] = (__bf16)(a[i] - (float)hi);
        }
#pragma unroll
        for (int j = 0; j < 4; ++j) {                // 4 N-tiles = full DOUT
            const v16bf bhi = *(const v16bf*)(bhbase + (size_t)(t * 4 + j) * 32 * 16);
            const v16bf blo = *(const v16bf*)(blbase + (size_t)(t * 4 + j) * 32 * 16);
            acc[j] = __builtin_amdgcn_wmma_f32_16x16x32_bf16(
                false, ahi, false, bhi, (short)0, acc[j], false, false);
            acc[j] = __builtin_amdgcn_wmma_f32_16x16x32_bf16(
                false, ahi, false, blo, (short)0, acc[j], false, false);
            acc[j] = __builtin_amdgcn_wmma_f32_16x16x32_bf16(
                false, alo, false, bhi, (short)0, acc[j], false, false);
        }
    }

    // D layout: VGPR i -> M = i + (lane<16 ? 0 : 8), N = 16*j + (lane&15)
    const int mo = (lane < 16) ? 0 : 8;
#pragma unroll
    for (int j = 0; j < 4; ++j) {
        int n = j * 16 + mrow;
        float bv = bias[h * DOUT_ + n];
#pragma unroll
        for (int i = 0; i < 8; ++i) {
            int rr = rbase + mo + i;
            if (rr < NN)
                feats[((size_t)h * NN + rr) * DOUT_ + n] = acc[j][i] + bv;
        }
    }
}

// ---------------------------------------------------------------------------
// Kernel 2: per-node attention scores a1/a2; init m=-inf, s=0; zero out row.
// One wave per node: lanes 0-15 -> head 0, lanes 16-31 -> head 1, 4 floats each.
// ---------------------------------------------------------------------------
__global__ __launch_bounds__(256)
void gat_scores(const float* __restrict__ feats, const float* __restrict__ a1w,
                const float* __restrict__ a1b, const float* __restrict__ a2w,
                const float* __restrict__ a2b, float* __restrict__ a1,
                float* __restrict__ a2, float* __restrict__ mbuf,
                float* __restrict__ sbuf, float* __restrict__ out) {
    int node = blockIdx.x * 8 + (threadIdx.x >> 5);
    if (node >= NN) return;                          // uniform per wave
    int lane = threadIdx.x & 31;
    int h  = lane >> 4;
    int c4 = (lane & 15) * 4;
    const float4 f  = *(const float4*)(feats + ((size_t)h * NN + node) * DOUT_ + c4);
    const float4 w1 = *(const float4*)(a1w + h * DOUT_ + c4);
    const float4 w2 = *(const float4*)(a2w + h * DOUT_ + c4);
    float p1 = f.x * w1.x + f.y * w1.y + f.z * w1.z + f.w * w1.w;
    float p2 = f.x * w2.x + f.y * w2.y + f.z * w2.z + f.w * w2.w;
#pragma unroll
    for (int m = 1; m < 16; m <<= 1) {               // reduce within 16-lane half
        p1 += __shfl_xor(p1, m);
        p2 += __shfl_xor(p2, m);
    }
    if ((lane & 15) == 0) {
        a1[(size_t)h * NN + node]   = p1 + a1b[h];
        a2[(size_t)h * NN + node]   = p2 + a2b[h];
        mbuf[(size_t)h * NN + node] = -__builtin_inff();
        sbuf[(size_t)h * NN + node] = 0.0f;
    }
    float4 z = {0.0f, 0.0f, 0.0f, 0.0f};
    *(float4*)(out + (size_t)node * (H_ * DOUT_) + lane * 4) = z;
}

// ---------------------------------------------------------------------------
// Kernel 3: edge logits + leaky-relu + segment max (atomics, unsorted rows)
// ---------------------------------------------------------------------------
__global__ void gat_edge_max(const int* __restrict__ idx, const float* __restrict__ a1,
                             const float* __restrict__ a2, float* __restrict__ vws,
                             float* __restrict__ mbuf) {
    int e = blockIdx.x * 256 + threadIdx.x;
    if (e >= NE) return;
    int r = idx[e], c = idx[NE + e];
#pragma unroll
    for (int h = 0; h < H_; ++h) {
        float v = a1[(size_t)h * NN + r] + a2[(size_t)h * NN + c];
        v = (v > 0.0f) ? v : 0.01f * v;              // leaky_relu slope 0.01
        vws[(size_t)h * NE + e] = v;
        atomicMaxF(mbuf + (size_t)h * NN + r, v);
    }
}

// ---------------------------------------------------------------------------
// Kernel 4: exp(v - rowmax) + segment sum
// ---------------------------------------------------------------------------
__global__ void gat_edge_exp(const int* __restrict__ idx, float* __restrict__ vws,
                             const float* __restrict__ mbuf, float* __restrict__ sbuf) {
    int e = blockIdx.x * 256 + threadIdx.x;
    if (e >= NE) return;
    int r = idx[e];
#pragma unroll
    for (int h = 0; h < H_; ++h) {
        float ex = expf(vws[(size_t)h * NE + e] - mbuf[(size_t)h * NN + r]);
        vws[(size_t)h * NE + e] = ex;
        atomicAdd(sbuf + (size_t)h * NN + r, ex);
    }
}

// ---------------------------------------------------------------------------
// Kernel 5: SpMM scatter: out[row, h*64+o] += (e/s) * feats[h][col][o]
// 64-lane group per (edge, head): coalesced 256B gather, f32 atomics scatter.
// ---------------------------------------------------------------------------
__global__ __launch_bounds__(256)
void gat_spmm(const int* __restrict__ idx, const float* __restrict__ vws,
              const float* __restrict__ sbuf, const float* __restrict__ feats,
              float* __restrict__ out) {
    int g = blockIdx.x * 4 + (threadIdx.x >> 6);     // group over (e, h)
    int o = threadIdx.x & 63;
    int e = g >> 1;
    int h = g & 1;
    if (e >= NE) return;
    int r = idx[e], c = idx[NE + e];
    float attn = vws[(size_t)h * NE + e] / sbuf[(size_t)h * NN + r];
    float val  = attn * feats[((size_t)h * NN + c) * DOUT_ + o];
    atomicAdd(out + (size_t)r * (H_ * DOUT_) + h * DOUT_ + o, val);
}

// ---------------------------------------------------------------------------
// Kernel 6: in-place ReLU on d_out
// ---------------------------------------------------------------------------
__global__ void gat_relu(float* __restrict__ out) {
    int t = blockIdx.x * 256 + threadIdx.x;          // over float4 elements
    float4* p = (float4*)out + t;
    float4 v = *p;
    v.x = fmaxf(v.x, 0.0f);
    v.y = fmaxf(v.y, 0.0f);
    v.z = fmaxf(v.z, 0.0f);
    v.w = fmaxf(v.w, 0.0f);
    *p = v;
}

// ---------------------------------------------------------------------------
extern "C" void kernel_launch(void* const* d_in, const int* in_sizes, int n_in,
                              void* d_out, int out_size, void* d_ws, size_t ws_size,
                              hipStream_t stream) {
    const float* features = (const float*)d_in[0];
    const int*   indices  = (const int*)d_in[1];
    const float* W        = (const float*)d_in[2];
    const float* b        = (const float*)d_in[3];
    const float* a1w      = (const float*)d_in[4];
    const float* a1b      = (const float*)d_in[5];
    const float* a2w      = (const float*)d_in[6];
    const float* a2b      = (const float*)d_in[7];
    float*       out      = (float*)d_out;

    char* ws = (char*)d_ws;
    float*  feats = (float*)(ws);                             // H*N*64 f32 = 25,600,000 B
    float*  a1    = (float*)(ws + 25600000);                  // H*N f32    =    400,000 B
    float*  a2    = (float*)(ws + 26000000);
    float*  mbuf  = (float*)(ws + 26400000);
    float*  sbuf  = (float*)(ws + 26800000);
    float*  vws   = (float*)(ws + 27200000);                  // H*E f32    =  6,400,000 B
    __bf16* whi   = (__bf16*)(ws + 33600000);                 // 2048*16 bf16 = 65,536 B
    __bf16* wlo   = (__bf16*)(ws + 33665536);

    // 0) pack W into WMMA bf16 hi/lo operand layout
    gat_pack_w<<<8, 256, 0, stream>>>(W, whi, wlo);
    // 1) dense projection via WMMA (split-bf16 for fp32-class accuracy)
    gat_gemm<<<dim3((NN + 127) / 128, H_), 256, 0, stream>>>(features, whi, wlo, b, feats);
    // 2) per-node scores + init m/s + zero output
    gat_scores<<<(NN + 7) / 8, 256, 0, stream>>>(feats, a1w, a1b, a2w, a2b,
                                                 a1, a2, mbuf, sbuf, out);
    // 3) edge logits + segment max
    gat_edge_max<<<(NE + 255) / 256, 256, 0, stream>>>(indices, a1, a2, vws, mbuf);
    // 4) exp + segment sum
    gat_edge_exp<<<(NE + 255) / 256, 256, 0, stream>>>(indices, vws, mbuf, sbuf);
    // 5) SpMM scatter with f32 atomics
    gat_spmm<<<(NE * H_ + 3) / 4, 256, 0, stream>>>(indices, vws, sbuf, feats, out);
    // 6) ReLU
    gat_relu<<<(NN * H_ * DOUT_ / 4 + 255) / 256, 256, 0, stream>>>(out);
}
